// LSTM1_44324062494743
// MI455X (gfx1250) — matrix operand order repaired and checked by reading, not compiled, not run on hardware
//
#include <hip/hip_runtime.h>
#include <hip/hip_bf16.h>

// ---------------------------------------------------------------------------
// LSTM, MI455X (gfx1250). Sizes: I=256, H=1024, O=256, S=512, B=64.
// Combined recurrence GEMM rows R=4H=4096 (gate-major), K = 256 + 1024 + 32:
//   kt 0..7   : x_s          (from pre-converted bf16 X, [s][b][i] layout)
//   kt 8..39  : h_{s-1}      (bf16, transposed [b][k] layout, K contiguous)
//   kt 40     : bias column  (constant panel: activation 1.0 then zeros)
// FC GEMM rows O=256, K = 1024 + 32 (kt 32 = bias panel).
// Activation tiles are 32(K) x 64(B) bf16, staged into padded LDS panels by
// the Tensor Data Mover (double-buffered), weights are pre-swizzled so each
// lane's WMMA A-fragment is one contiguous 32-byte load.
// ---------------------------------------------------------------------------

#define I_DIM 256
#define H_DIM 1024
#define O_DIM 256
#define S_LEN 512
#define B_SZ  64

#define KT_REC 41          // 1312/32 (last tile = bias)
#define KT_FC  33          // 1056/32 (last tile = bias)
#define NWGS   16          // persistent workgroups

typedef __bf16          bf16x16 __attribute__((ext_vector_type(16)));
typedef float           f32x8   __attribute__((ext_vector_type(8)));
typedef unsigned short  u16x8   __attribute__((ext_vector_type(8)));
typedef unsigned int    u32x4   __attribute__((ext_vector_type(4)));
typedef int             i32x8   __attribute__((ext_vector_type(8)));
typedef int             i32x4   __attribute__((ext_vector_type(4)));

#if __has_builtin(__builtin_amdgcn_tensor_load_to_lds) && \
    __has_builtin(__builtin_amdgcn_s_wait_tensorcnt)
#define USE_TDM 1
#else
#define USE_TDM 0
#endif

static __device__ __forceinline__ unsigned short f2bf(float f) {
    unsigned u = __builtin_bit_cast(unsigned, f);
    unsigned r = u + 0x7FFFu + ((u >> 16) & 1u);   // round-to-nearest-even
    return (unsigned short)(r >> 16);
}
static __device__ __forceinline__ float sigmoidf_fast(float x) {
    return 1.0f / (1.0f + __expf(-x));
}
static __device__ __forceinline__ float tanhf_fast(float x) {
    return 2.0f / (1.0f + __expf(-2.0f * x)) - 1.0f;
}

#if USE_TDM
// Tensor DMA: load a 32(K) x 64(rows) bf16 tile, row stride `strideElems`,
// into LDS at `lds_off` with 8-DWORD padding every 16 DWORDs (-> [64][48] panel).
// D# per CDNA5 ISA section 8 (2D tensor: remaining groups zero).
// 6-arg builtin form (clang-23 / therock signature).
static __device__ __forceinline__ void tdm_load_tile(unsigned lds_off,
                                                     const void* gptr,
                                                     unsigned strideElems) {
    unsigned long long ga = (unsigned long long)(size_t)gptr;
    u32x4 g0;
    g0[0] = 1u;                                    // count=1, user descriptor
    g0[1] = lds_off;                               // lds_addr (bytes)
    g0[2] = (unsigned)ga;                          // global_addr[31:0]
    g0[3] = (unsigned)(ga >> 32) | (2u << 30);     // global_addr[56:32] | type=2
    i32x8 g1;
    g1[0] = (1 << 16)      // data_size = 2 bytes
          | (1 << 20)      // pad_enable
          | (3 << 22)      // pad_interval: every 16 DWORDs (row = 32 bf16)
          | (7 << 25);     // pad_amount: 8 DWORDs (panel row stride 48 bf16)
    g1[1] = (32 << 16);    // tensor_dim0 = 32
    g1[2] = (64 << 16);    // tensor_dim1 = 64
    g1[3] = (32 << 16);    // tile_dim0 = 32
    g1[4] = 64;            // tile_dim1 = 64, tile_dim2 = 0
    g1[5] = (int)strideElems;  // tensor_dim0_stride (data_size units)
    g1[6] = 0;
    g1[7] = 0;
    i32x4 z4 = {0, 0, 0, 0};
    i32x8 z8 = {0, 0, 0, 0, 0, 0, 0, 0};
    __builtin_amdgcn_tensor_load_to_lds(g0, g1, z4, z4, z8, 0);
}
#endif

// Manual staging fallback: copy 32x64 bf16 tile into padded panel.
// 256 threads: thread t -> row b = t>>2, 8 contiguous k at (t&3)*8 (one 16B load).
static __device__ __forceinline__ void stage_tile(unsigned short (*panel)[48],
                                                  const unsigned short* src,
                                                  unsigned strideElems, int tid) {
    int b   = tid >> 2;
    int kq0 = (tid & 3) * 8;
    u16x8 v = *reinterpret_cast<const u16x8*>(src + (size_t)b * strideElems + kq0);
    *reinterpret_cast<u16x8*>(&panel[b][kq0]) = v;
}

// ---------------------------------------------------------------------------
// One-shot conversions
// ---------------------------------------------------------------------------

// X [I][S][B] fp32  ->  Xbf [S][B][I] bf16 (K-contiguous for B-fragments/TDM)
__global__ void conv_x(const float* __restrict__ X, unsigned short* __restrict__ Xbf,
                       long long n) {
    long long e = (long long)blockIdx.x * blockDim.x + threadIdx.x;
    if (e >= n) return;
    int i = (int)(e & (I_DIM - 1));
    long long sb = e >> 8;
    int b = (int)(sb & (B_SZ - 1));
    int s = (int)(sb >> 6);
    Xbf[e] = f2bf(X[((size_t)i * S_LEN + s) * B_SZ + b]);
}

// A-fragment swizzle for WMMA 16-bit A 16x32 (contiguous k order used here, not the ISA table order):
// lanes 0-15: M=lane, K=0..15; lanes 16-31: M=lane-16, K=16..31 (matches contiguous B)
static __device__ __forceinline__ int a_klocal(int j, int lane) {
    return j + ((lane >= 16) ? 16 : 0);
}

__global__ void conv_rec_weights(const float* __restrict__ Wf, const float* __restrict__ Wi,
                                 const float* __restrict__ Wo, const float* __restrict__ Wc,
                                 const float* __restrict__ Uf, const float* __restrict__ Ui,
                                 const float* __restrict__ Uo, const float* __restrict__ Uc,
                                 const float* __restrict__ bf, const float* __restrict__ bi,
                                 const float* __restrict__ bo, const float* __restrict__ bc,
                                 unsigned short* __restrict__ dst, long long n) {
    long long e = (long long)blockIdx.x * blockDim.x + threadIdx.x;
    if (e >= n) return;
    int j    = (int)(e & 15);
    int lane = (int)((e >> 4) & 31);
    long long rem = e >> 9;
    int kt = (int)(rem % KT_REC);
    int tr = (int)(rem / KT_REC);
    int row  = tr * 16 + (lane & 15);
    int g    = row >> 10;
    int hrow = row & 1023;
    int k = kt * 32 + a_klocal(j, lane);
    float val;
    if (k < I_DIM) {
        const float* W = (g == 0) ? Wf : (g == 1) ? Wi : (g == 2) ? Wo : Wc;
        val = W[hrow * I_DIM + k];
    } else if (k < I_DIM + H_DIM) {
        const float* U = (g == 0) ? Uf : (g == 1) ? Ui : (g == 2) ? Uo : Uc;
        val = U[hrow * H_DIM + (k - I_DIM)];
    } else if (k == I_DIM + H_DIM) {
        const float* B = (g == 0) ? bf : (g == 1) ? bi : (g == 2) ? bo : bc;
        val = B[hrow];
    } else {
        val = 0.0f;
    }
    dst[e] = f2bf(val);
}

__global__ void conv_fc_weights(const float* __restrict__ fc_w, const float* __restrict__ fc_b,
                                unsigned short* __restrict__ dst, long long n) {
    long long e = (long long)blockIdx.x * blockDim.x + threadIdx.x;
    if (e >= n) return;
    int j    = (int)(e & 15);
    int lane = (int)((e >> 4) & 31);
    long long rem = e >> 9;
    int kt = (int)(rem % KT_FC);
    int tr = (int)(rem / KT_FC);
    int row = tr * 16 + (lane & 15);
    int k = kt * 32 + a_klocal(j, lane);
    float val;
    if (k < H_DIM)       val = fc_w[row * H_DIM + k];
    else if (k == H_DIM) val = fc_b[row];
    else                 val = 0.0f;
    dst[e] = f2bf(val);
}

__global__ void init_state(unsigned short* __restrict__ hbufT, int* __restrict__ bar) {
    int idx = blockIdx.x * blockDim.x + threadIdx.x;
    if (idx < H_DIM * B_SZ) hbufT[idx] = 0;     // h0 = 0 (parity-0, [b][k])
    if (idx < 2) bar[idx] = 0;
}

// ---------------------------------------------------------------------------
// Device-scope sense barrier across the 16 persistent workgroups.
// ---------------------------------------------------------------------------
static __device__ __forceinline__ void grid_barrier(int* bar) {
    __syncthreads();
    if (threadIdx.x == 0) {
        int gen = __hip_atomic_load(&bar[1], __ATOMIC_ACQUIRE, __HIP_MEMORY_SCOPE_AGENT);
        int a = __hip_atomic_fetch_add(&bar[0], 1, __ATOMIC_ACQ_REL, __HIP_MEMORY_SCOPE_AGENT);
        if (a == NWGS - 1) {
            __hip_atomic_store(&bar[0], 0, __ATOMIC_RELAXED, __HIP_MEMORY_SCOPE_AGENT);
            __hip_atomic_fetch_add(&bar[1], 1, __ATOMIC_RELEASE, __HIP_MEMORY_SCOPE_AGENT);
        } else {
            while (__hip_atomic_load(&bar[1], __ATOMIC_ACQUIRE, __HIP_MEMORY_SCOPE_AGENT) == gen) {
                __builtin_amdgcn_s_sleep(1);
            }
        }
    }
    __syncthreads();
}

// ---------------------------------------------------------------------------
// Persistent LSTM recurrence: 16 WGs x 256 threads (8 wave32).
// WG owns 64 h-rows x 4 gates; wave w: sub-chunk sc=w&3, batch half bh=w>>2.
// c lives in registers for all 512 steps. Activation tiles staged by TDM.
// ---------------------------------------------------------------------------
__global__ __launch_bounds__(256, 1)
void lstm_recurrence(const unsigned short* __restrict__ Xbf,     // [S][B][I]
                     const unsigned short* __restrict__ swW,
                     unsigned short* __restrict__ hbufT,          // 2 x [B][H]
                     unsigned short* __restrict__ hsAll,          // [S][B][H]
                     float* __restrict__ outH, float* __restrict__ outC,
                     int* __restrict__ bar) {
    const int wg   = blockIdx.x;
    const int tid  = threadIdx.x;
    const int w    = tid >> 5;
    const int lane = tid & 31;
    const int sc   = w & 3;
    const int bh   = w >> 2;
    const int n    = lane & 15;
    const int hi   = lane >> 4;

    __shared__ __attribute__((aligned(32))) unsigned short panel[2][B_SZ][48];
    __shared__ __attribute__((aligned(32))) unsigned short panelBias[B_SZ][48];

    // constant bias panel: k==1280 column -> 1.0, rest of tile -> 0
    for (int idx = tid; idx < B_SZ * 48; idx += 256)
        panelBias[idx / 48][idx % 48] = ((idx % 48) == 0) ? 0x3F80 : 0;

    float creg[2][8];
    #pragma unroll
    for (int ct = 0; ct < 2; ++ct)
        #pragma unroll
        for (int v = 0; v < 8; ++v) creg[ct][v] = 0.0f;

#if USE_TDM
    const unsigned lds0 = (unsigned)(size_t)(void*)&panel[0][0][0];
    const unsigned lds1 = (unsigned)(size_t)(void*)&panel[1][0][0];
#endif
    __syncthreads();

    for (int s = 0; s < S_LEN; ++s) {
        const int par = s & 1;
        const unsigned short* hT  = hbufT + (size_t)par * (B_SZ * H_DIM);
        const unsigned short* xS  = Xbf + (size_t)s * (B_SZ * I_DIM);

        f32x8 acc[4][2];
        #pragma unroll
        for (int g = 0; g < 4; ++g) { acc[g][0] = {}; acc[g][1] = {}; }

#if USE_TDM
        if (w == 0) tdm_load_tile(lds0, xS, I_DIM);   // tile kt=0
#endif
        for (int kt = 0; kt < KT_REC; ++kt) {
            const unsigned short (*pan)[48];
            if (kt < KT_REC - 1) {
                const int cur = kt & 1;
#if USE_TDM
                if (w == 0) {
                    if (kt + 1 < KT_REC - 1) {
                        int nk = kt + 1;
                        if (nk < 8) tdm_load_tile(cur ? lds0 : lds1, xS + nk * 32, I_DIM);
                        else        tdm_load_tile(cur ? lds0 : lds1, hT + (nk - 8) * 32, H_DIM);
                        __builtin_amdgcn_s_wait_tensorcnt(1);   // tile kt retired
                    } else {
                        __builtin_amdgcn_s_wait_tensorcnt(0);   // drain
                    }
                }
#else
                if (kt < 8) stage_tile(panel[cur], xS + kt * 32, I_DIM, tid);
                else        stage_tile(panel[cur], hT + (kt - 8) * 32, H_DIM, tid);
#endif
                __syncthreads();
                pan = panel[cur];
            } else {
                pan = panelBias;
            }

            // B fragments: column b, K contiguous, 32B-aligned LDS reads
            bf16x16 bfr0 = *reinterpret_cast<const bf16x16*>(&pan[bh * 32 + n][hi * 16]);
            bf16x16 bfr1 = *reinterpret_cast<const bf16x16*>(&pan[bh * 32 + 16 + n][hi * 16]);

            if (kt + 1 < KT_REC)
                __builtin_prefetch(swW + (((size_t)(wg * 4 + sc) * KT_REC + kt + 1) * 32 + lane) * 16, 0, 0);

            #pragma unroll
            for (int g = 0; g < 4; ++g) {
                const unsigned short* ap =
                    swW + (((size_t)(g * 64 + wg * 4 + sc) * KT_REC + kt) * 32 + lane) * 16;
                bf16x16 af = *reinterpret_cast<const bf16x16*>(ap);
                acc[g][0] = __builtin_amdgcn_wmma_f32_16x16x32_bf16(
                    false, af, false, bfr0, (short)0, acc[g][0], false, false);
                acc[g][1] = __builtin_amdgcn_wmma_f32_16x16x32_bf16(
                    false, af, false, bfr1, (short)0, acc[g][1], false, false);
            }
            if (kt < KT_REC - 1) __syncthreads();   // release panel before reuse
        }

        // gates + state. C/D layout: M = v + hi*8, N = lane&15
        const int hrow_base = wg * 64 + sc * 16 + hi * 8;
        const size_t nextT = (size_t)(par ^ 1) * (B_SZ * H_DIM);
        #pragma unroll
        for (int ct = 0; ct < 2; ++ct) {
            const int b = bh * 32 + ct * 16 + n;
            u16x8 hv;
            float hval[8], cval[8];
            #pragma unroll
            for (int v = 0; v < 8; ++v) {
                float fg = sigmoidf_fast(acc[0][ct][v]);
                float ig = sigmoidf_fast(acc[1][ct][v]);
                float og = sigmoidf_fast(acc[2][ct][v]);
                float ch = tanhf_fast(acc[3][ct][v]);
                float c  = fg * creg[ct][v] + ig * ch;
                creg[ct][v] = c;
                float h  = og * tanhf_fast(c);
                hval[v] = h; cval[v] = c;
                hv[v] = f2bf(h);
            }
            // packed 16B stores ([b][k] layouts, hrow_base 8-aligned)
            *reinterpret_cast<u16x8*>(&hbufT[nextT + (size_t)b * H_DIM + hrow_base]) = hv;
            *reinterpret_cast<u16x8*>(&hsAll[((size_t)s * B_SZ + b) * H_DIM + hrow_base]) = hv;
            if (s == S_LEN - 1) {
                #pragma unroll
                for (int v = 0; v < 8; ++v) {
                    outH[(hrow_base + v) * B_SZ + b] = hval[v];
                    outC[(hrow_base + v) * B_SZ + b] = cval[v];
                }
            }
        }
        __threadfence();
        grid_barrier(bar);
    }
}

// ---------------------------------------------------------------------------
// Final: out[o,s,b] = fc_w @ h_s + fc_b, softmax over o. One block per s.
// ---------------------------------------------------------------------------
__global__ __launch_bounds__(256, 2)
void fc_softmax(const unsigned short* __restrict__ swFC,
                const unsigned short* __restrict__ hsAll,   // [S][B][H]
                float* __restrict__ P) {
    const int s    = blockIdx.x;
    const int tid  = threadIdx.x;
    const int w    = tid >> 5;
    const int lane = tid & 31;
    const int n    = lane & 15;
    const int hi   = lane >> 4;

    __shared__ __attribute__((aligned(32))) unsigned short panel[2][B_SZ][48];
    __shared__ __attribute__((aligned(32))) unsigned short panelBias[B_SZ][48];
    __shared__ float red[8][B_SZ];
    __shared__ float gred[B_SZ];

    for (int idx = tid; idx < B_SZ * 48; idx += 256)
        panelBias[idx / 48][idx % 48] = ((idx % 48) == 0) ? 0x3F80 : 0;

    f32x8 acc[2][4];
    #pragma unroll
    for (int rt = 0; rt < 2; ++rt)
        #pragma unroll
        for (int ct = 0; ct < 4; ++ct) acc[rt][ct] = {};

    const unsigned short* hs_s = hsAll + (size_t)s * B_SZ * H_DIM;
#if USE_TDM
    const unsigned lds0 = (unsigned)(size_t)(void*)&panel[0][0][0];
    const unsigned lds1 = (unsigned)(size_t)(void*)&panel[1][0][0];
#endif
    __syncthreads();
#if USE_TDM
    if (w == 0) tdm_load_tile(lds0, hs_s, H_DIM);
#endif

    for (int kt = 0; kt < KT_FC; ++kt) {
        const unsigned short (*pan)[48];
        if (kt < KT_FC - 1) {
            const int cur = kt & 1;
#if USE_TDM
            if (w == 0) {
                if (kt + 1 < KT_FC - 1) {
                    tdm_load_tile(cur ? lds0 : lds1, hs_s + (kt + 1) * 32, H_DIM);
                    __builtin_amdgcn_s_wait_tensorcnt(1);   // tile kt retired
                } else {
                    __builtin_amdgcn_s_wait_tensorcnt(0);   // drain
                }
            }
#else
            stage_tile(panel[cur], hs_s + kt * 32, H_DIM, tid);
#endif
            __syncthreads();
            pan = panel[cur];
        } else {
            pan = panelBias;
        }

        bf16x16 bfr[4];
        #pragma unroll
        for (int ct = 0; ct < 4; ++ct)
            bfr[ct] = *reinterpret_cast<const bf16x16*>(&pan[ct * 16 + n][hi * 16]);

        #pragma unroll
        for (int rt = 0; rt < 2; ++rt) {
            const unsigned short* ap =
                swFC + (((size_t)(2 * w + rt) * KT_FC + kt) * 32 + lane) * 16;
            bf16x16 af = *reinterpret_cast<const bf16x16*>(ap);
            #pragma unroll
            for (int ct = 0; ct < 4; ++ct)
                acc[rt][ct] = __builtin_amdgcn_wmma_f32_16x16x32_bf16(
                    false, af, false, bfr[ct], (short)0, acc[rt][ct], false, false);
        }
        if (kt < KT_FC - 1) __syncthreads();
    }

    // softmax over o (256 rows across 8 waves), per column b
    float m[4];
    #pragma unroll
    for (int ct = 0; ct < 4; ++ct) {
        m[ct] = -3.4e38f;
        #pragma unroll
        for (int rt = 0; rt < 2; ++rt)
            #pragma unroll
            for (int v = 0; v < 8; ++v) m[ct] = fmaxf(m[ct], acc[rt][ct][v]);
        m[ct] = fmaxf(m[ct], __shfl_xor(m[ct], 16, 32));
    }
    if (lane < 16) {
        #pragma unroll
        for (int ct = 0; ct < 4; ++ct) red[w][ct * 16 + lane] = m[ct];
    }
    __syncthreads();
    if (tid < B_SZ) {
        float g = -3.4e38f;
        #pragma unroll
        for (int ww = 0; ww < 8; ++ww) g = fmaxf(g, red[ww][tid]);
        gred[tid] = g;
    }
    __syncthreads();

    float ssum[4];
    #pragma unroll
    for (int ct = 0; ct < 4; ++ct) {
        float gm = gred[ct * 16 + n];
        float acc_s = 0.0f;
        #pragma unroll
        for (int rt = 0; rt < 2; ++rt)
            #pragma unroll
            for (int v = 0; v < 8; ++v) {
                float e = __expf(acc[rt][ct][v] - gm);
                acc[rt][ct][v] = e;
                acc_s += e;
            }
        ssum[ct] = acc_s + __shfl_xor(acc_s, 16, 32);
    }
    __syncthreads();
    if (lane < 16) {
        #pragma unroll
        for (int ct = 0; ct < 4; ++ct) red[w][ct * 16 + lane] = ssum[ct];
    }
    __syncthreads();
    if (tid < B_SZ) {
        float g = 0.0f;
        #pragma unroll
        for (int ww = 0; ww < 8; ++ww) g += red[ww][tid];
        gred[tid] = g;
    }
    __syncthreads();

    #pragma unroll
    for (int ct = 0; ct < 4; ++ct) {
        const int b = ct * 16 + n;
        float inv = 1.0f / gred[b];
        #pragma unroll
        for (int rt = 0; rt < 2; ++rt)
            #pragma unroll
            for (int v = 0; v < 8; ++v) {
                int o = (2 * w + rt) * 16 + hi * 8 + v;
                P[((size_t)o * S_LEN + s) * B_SZ + b] = acc[rt][ct][v] * inv;
            }
    }
}

// ---------------------------------------------------------------------------
// Host launcher
// ---------------------------------------------------------------------------
extern "C" void kernel_launch(void* const* d_in, const int* in_sizes, int n_in,
                              void* d_out, int out_size, void* d_ws, size_t ws_size,
                              hipStream_t stream) {
    (void)in_sizes; (void)n_in; (void)out_size; (void)ws_size;

    const float* X    = (const float*)d_in[0];
    const float* Wf   = (const float*)d_in[1];
    const float* Wi   = (const float*)d_in[2];
    const float* Wo   = (const float*)d_in[3];
    const float* Wc   = (const float*)d_in[4];
    const float* Uf   = (const float*)d_in[5];
    const float* Ui   = (const float*)d_in[6];
    const float* Uo   = (const float*)d_in[7];
    const float* Uc   = (const float*)d_in[8];
    const float* bf   = (const float*)d_in[9];
    const float* bi   = (const float*)d_in[10];
    const float* bo   = (const float*)d_in[11];
    const float* bc   = (const float*)d_in[12];
    const float* fc_w = (const float*)d_in[13];
    const float* fc_b = (const float*)d_in[14];

    float* out  = (float*)d_out;
    float* outP = out;                                        // [O][S][B]
    float* outH = out + (size_t)O_DIM * S_LEN * B_SZ;         // [H][B]
    float* outC = outH + (size_t)H_DIM * B_SZ;                // [H][B]

    const long long NW  = 256LL * KT_REC * 32 * 16;           // swizzled rec weights
    const long long NFC = 16LL  * KT_FC  * 32 * 16;           // swizzled fc weights
    const long long NX  = (long long)S_LEN * B_SZ * I_DIM;    // Xbf elements

    char* ws = (char*)d_ws;
    size_t off = 0;
    auto take = [&](size_t bytes) { size_t r = off; off = (off + bytes + 255) & ~(size_t)255; return r; };
    unsigned short* swW   = (unsigned short*)(ws + take((size_t)NW * 2));
    unsigned short* swFC  = (unsigned short*)(ws + take((size_t)NFC * 2));
    unsigned short* Xbf   = (unsigned short*)(ws + take((size_t)NX * 2));
    unsigned short* hbufT = (unsigned short*)(ws + take((size_t)2 * B_SZ * H_DIM * 2));
    unsigned short* hsAll = (unsigned short*)(ws + take((size_t)S_LEN * B_SZ * H_DIM * 2));
    int* bar = (int*)(ws + take(2 * sizeof(int)));

    init_state<<<(H_DIM * B_SZ + 255) / 256, 256, 0, stream>>>(hbufT, bar);
    conv_x<<<(unsigned)((NX + 255) / 256), 256, 0, stream>>>(X, Xbf, NX);
    conv_rec_weights<<<(unsigned)((NW + 255) / 256), 256, 0, stream>>>(
        Wf, Wi, Wo, Wc, Uf, Ui, Uo, Uc, bf, bi, bo, bc, swW, NW);
    conv_fc_weights<<<(unsigned)((NFC + 255) / 256), 256, 0, stream>>>(fc_w, fc_b, swFC, NFC);

    lstm_recurrence<<<NWGS, 256, 0, stream>>>(Xbf, swW, hbufT, hsAll, outH, outC, bar);

    fc_softmax<<<S_LEN, 256, 0, stream>>>(swFC, hsAll, outP);
}
